// LemInFrame_85744727097797
// MI455X (gfx1250) — compile-verified, matrix operand order checked by reading, not run on hardware
//
#include <hip/hip_runtime.h>
#include <hip/hip_bf16.h>
#include <cstdint>
#include <cstddef>

#define E_EDGES 800000
#define N_NODES 50000
#define OH 128
#define LATF 128
#define NB 8
#define RMAXF 5.0f
#define EPSF 1e-8f

typedef _Float16 h4  __attribute__((ext_vector_type(4)));
typedef _Float16 h8  __attribute__((ext_vector_type(8)));
typedef _Float16 h16 __attribute__((ext_vector_type(16)));
typedef float    f8  __attribute__((ext_vector_type(8)));

// LDS layout (in halves)
#define W1T_OFF 0           // [128][160]  (K padded 136->160, zeros)
#define W2T_OFF 20480       // [128][128]
#define W3T_OFF 36864       // [128][128]
#define WET_OFF 53248       // [96][128]
#define STG_OFF 65536       // 8 waves x [16][160]
#define STG_STRIDE 2560
#define AUX_OFF (STG_OFF + 8 * STG_STRIDE)   // aux floats after (8 waves x 192 f32)
#define SHMEM_BYTES (AUX_OFF * 2 + 8 * 192 * 4)   // 178176 B

__device__ __forceinline__ h16 load_frag(const _Float16* base, int stride, int row, int k0) {
  // A/B 16-bit fragment: halves 0..7 = K[k0..k0+7], halves 8..15 = K[k0+16..k0+23]
  const h8 lo = *(const h8*)(base + row * stride + k0);
  const h8 hi = *(const h8*)(base + row * stride + k0 + 16);
  return __builtin_shufflevector(lo, hi, 0,1,2,3,4,5,6,7,8,9,10,11,12,13,14,15);
}

__device__ __forceinline__ f8 wmma16(h16 a, h16 b, f8 c) {
  return __builtin_amdgcn_wmma_f32_16x16x32_f16(false, a, false, b, (short)0, c, false, false);
}

__device__ __forceinline__ float silu(float x) { return x / (1.0f + __expf(-x)); }

// ---------------------------------------------------------------------------
// Kernel 0: weights -> f16, transposed (Wt[n][k]) into workspace, W1 K-padded.
// ---------------------------------------------------------------------------
__global__ __launch_bounds__(256)
void prep_weights(const float* __restrict__ W1, const float* __restrict__ W2,
                  const float* __restrict__ W3, const float* __restrict__ We,
                  _Float16* __restrict__ wts) {
  const int t = blockIdx.x * 256 + threadIdx.x;
  if (t >= 65536) return;
  if (t < 20480) {                    // W1t [128][160]
    const int n = t / 160, k = t % 160;
    wts[t] = (k < 136) ? (_Float16)W1[k * 128 + n] : (_Float16)0.0f;
  } else if (t < 36864) {             // W2t [128][128]
    const int u = t - 20480, n = u / 128, k = u % 128;
    wts[t] = (_Float16)W2[k * 128 + n];
  } else if (t < 53248) {             // W3t [128][128]
    const int u = t - 36864, n = u / 128, k = u % 128;
    wts[t] = (_Float16)W3[k * 128 + n];
  } else {                            // W_envt [96][128]
    const int u = t - 53248, n = u / 128, k = u % 128;
    wts[t] = (_Float16)We[k * 96 + n];
  }
}

// ---------------------------------------------------------------------------
// Kernel 1: fully fused edge pipeline (one 16-edge tile per wave).
// ---------------------------------------------------------------------------
__global__ __launch_bounds__(256)
void edge_mlp_wmma(const int*   __restrict__ edge_index,
                   const float* __restrict__ edge_sh,
                   const float* __restrict__ edge_length,
                   const float* __restrict__ edge_one_hot,
                   const float* __restrict__ bessel_w,
                   const _Float16* __restrict__ wts,
                   float* __restrict__ out_lat,
                   float* __restrict__ node_acc,
                   float* __restrict__ out_edgef,
                   float* __restrict__ out_cut) {
  extern __shared__ _Float16 lds[];
  const int tid = threadIdx.x;
  const int wv = tid >> 5;
  const int lane = tid & 31;
  const int nlane = lane & 15;
  const int halfsel = lane >> 4;
  const int mbase = halfsel * 8;   // C/D rows: VGPR r -> M = r + mbase
  const int kofs  = halfsel * 8;   // A/B K sub-offset per lane half

  // ---- cooperative copy of 128KB f16 weights into LDS ----
  {
    const uint4* src = (const uint4*)wts;
    uint4* dst = (uint4*)lds;
    #pragma unroll
    for (int i = 0; i < 32; ++i) dst[i * 256 + tid] = src[i * 256 + tid];
  }

  _Float16* stg = lds + STG_OFF + wv * STG_STRIDE;            // [16][160] f16
  float* aux    = ((float*)(lds + AUX_OFF)) + wv * 192;
  float* auxcut = aux;          // [16]
  float* auxsh  = aux + 16;     // [16][9]
  int*   auxec  = (int*)(aux + 160);   // [16]

  const int tile = blockIdx.x * 8 + wv;
  const int e0 = tile * 16;

  // keep HBM busy: prefetch a future tile's one-hot block
  __builtin_prefetch(edge_one_hot + (size_t)(e0 + 128) * OH + lane * 16, 0, 1);

  // ---- stage A0 = [one_hot(128) | bessel(8) | zeros(24)] as f16 ----
  #pragma unroll
  for (int i = 0; i < 16; ++i) {
    const int idx = i * 32 + lane;
    const int row = idx >> 5;
    const int c4 = idx & 31;
    const float4 v = ((const float4*)(edge_one_hot + (size_t)(e0 + row) * OH))[c4];
    h4 hv; hv[0] = (_Float16)v.x; hv[1] = (_Float16)v.y;
    hv[2] = (_Float16)v.z; hv[3] = (_Float16)v.w;
    *(h4*)(stg + row * 160 + c4 * 4) = hv;
  }
  if (lane < 16) {
    const int e = e0 + lane;
    const float r = edge_length[e];
    const float rs = 2.0f / (RMAXF * r);
    #pragma unroll
    for (int j = 0; j < NB; ++j)
      stg[lane * 160 + 128 + j] = (_Float16)(rs * __sinf(bessel_w[j] * r));
    const h8 z = {};
    *(h8*)(stg + lane * 160 + 136) = z;
    *(h8*)(stg + lane * 160 + 144) = z;
    *(h8*)(stg + lane * 160 + 152) = z;
    // polynomial cutoff, P=6: 1 - 28 x^6 + 48 x^7 - 21 x^8
    const float x = r * (1.0f / RMAXF);
    const float x2 = x * x, x6 = x2 * x2 * x2;
    float c = 1.0f - 28.0f * x6 + 48.0f * x6 * x - 21.0f * x6 * x2;
    c = (x < 1.0f) ? c : 0.0f;
    out_cut[e] = c;
    auxcut[lane] = (c > 0.0f) ? c : 0.0f;
    auxec[lane] = edge_index[e];             // edge_center = row 0 of (2,E)
    #pragma unroll
    for (int j = 0; j < 9; ++j) auxsh[lane * 9 + j] = edge_sh[(size_t)e * 9 + j];
  }
  __syncthreads();

  const _Float16* W1t = lds + W1T_OFF;
  const _Float16* W2t = lds + W2T_OFF;
  const _Float16* W3t = lds + W3T_OFF;
  const _Float16* Wet = lds + WET_OFF;

  f8 h[8];

  // ---- GEMM1: (16x160) @ (160x128), silu ----
  #pragma unroll
  for (int nt = 0; nt < 8; ++nt) {
    f8 acc = {};
    #pragma unroll
    for (int kc = 0; kc < 5; ++kc) {
      const int k0 = kc * 32 + kofs;
      acc = wmma16(load_frag(stg, 160, nlane, k0),
                   load_frag(W1t, 160, nt * 16 + nlane, k0), acc);
    }
    #pragma unroll
    for (int r = 0; r < 8; ++r) acc[r] = silu(acc[r]);
    h[nt] = acc;
  }
  __syncthreads();
  #pragma unroll
  for (int nt = 0; nt < 8; ++nt)
    #pragma unroll
    for (int r = 0; r < 8; ++r)
      stg[(mbase + r) * 160 + nt * 16 + nlane] = (_Float16)h[nt][r];
  __syncthreads();

  // ---- GEMM2: (16x128) @ (128x128), silu ----
  #pragma unroll
  for (int nt = 0; nt < 8; ++nt) {
    f8 acc = {};
    #pragma unroll
    for (int kc = 0; kc < 4; ++kc) {
      const int k0 = kc * 32 + kofs;
      acc = wmma16(load_frag(stg, 160, nlane, k0),
                   load_frag(W2t, 128, nt * 16 + nlane, k0), acc);
    }
    #pragma unroll
    for (int r = 0; r < 8; ++r) acc[r] = silu(acc[r]);
    h[nt] = acc;
  }
  __syncthreads();
  #pragma unroll
  for (int nt = 0; nt < 8; ++nt)
    #pragma unroll
    for (int r = 0; r < 8; ++r)
      stg[(mbase + r) * 160 + nt * 16 + nlane] = (_Float16)h[nt][r];
  __syncthreads();

  // ---- GEMM3: latents = cutoff * (h @ W3), write out + restage ----
  float cm[8];
  {
    const float4 ca = *(const float4*)(auxcut + mbase);
    const float4 cb = *(const float4*)(auxcut + mbase + 4);
    cm[0] = ca.x; cm[1] = ca.y; cm[2] = ca.z; cm[3] = ca.w;
    cm[4] = cb.x; cm[5] = cb.y; cm[6] = cb.z; cm[7] = cb.w;
  }
  #pragma unroll
  for (int nt = 0; nt < 8; ++nt) {
    f8 acc = {};
    #pragma unroll
    for (int kc = 0; kc < 4; ++kc) {
      const int k0 = kc * 32 + kofs;
      acc = wmma16(load_frag(stg, 160, nlane, k0),
                   load_frag(W3t, 128, nt * 16 + nlane, k0), acc);
    }
    #pragma unroll
    for (int r = 0; r < 8; ++r) acc[r] *= cm[r];
    h[nt] = acc;
  }
  __syncthreads();
  #pragma unroll
  for (int nt = 0; nt < 8; ++nt)
    #pragma unroll
    for (int r = 0; r < 8; ++r) {
      const float lv = h[nt][r];
      out_lat[(size_t)(e0 + mbase + r) * LATF + nt * 16 + nlane] = lv;
      stg[(mbase + r) * 160 + nt * 16 + nlane] = (_Float16)lv;
    }
  __syncthreads();

  // ---- GEMM4: w = latents @ W_env, (16x128)@(128x96) ----
  f8 w[6];
  #pragma unroll
  for (int nt = 0; nt < 6; ++nt) {
    f8 acc = {};
    #pragma unroll
    for (int kc = 0; kc < 4; ++kc) {
      const int k0 = kc * 32 + kofs;
      acc = wmma16(load_frag(stg, 160, nlane, k0),
                   load_frag(Wet, 128, nt * 16 + nlane, k0), acc);
    }
    w[nt] = acc;
  }

  // ---- edge features + L2-resident segment-sum atomics ----
  #pragma unroll
  for (int nt = 0; nt < 6; ++nt) {
    const int j = nt * 16 + nlane;        // 0..95
    const int g = j >> 5;                 // 0,1,2 (uniform per nt)
    const int mi = j & 31;
    const int shoff = (g == 0) ? 0 : ((g == 1) ? 1 : 4);
    const int dim   = (g == 0) ? 1 : ((g == 1) ? 3 : 5);
    const int colb  = (g == 0) ? mi : ((g == 1) ? (32 + mi * 3) : (128 + mi * 5));
    #pragma unroll
    for (int r = 0; r < 8; ++r) {
      const int m = mbase + r;
      const int e = e0 + m;
      const float wval = w[nt][r];
      float* orow = out_edgef + (size_t)e * 288 + colb;
      float* arow = node_acc + (size_t)auxec[m] * 288 + colb;
      for (int c = 0; c < dim; ++c) {
        const float f = wval * auxsh[m * 9 + shoff + c];
        orow[c] = f;
        atomicAdd(arow + c, f);
      }
    }
  }
}

// ---------------------------------------------------------------------------
// Kernel 2: in-place node layernorms (one wave32 per node).
// ---------------------------------------------------------------------------
__global__ __launch_bounds__(256)
void node_norm(float* __restrict__ node, const float* __restrict__ ln_w0,
               const float* __restrict__ ln_b0, const float* __restrict__ ln_w1) {
  const int wv = threadIdx.x >> 5;
  const int lane = threadIdx.x & 31;
  const int n = blockIdx.x * 8 + wv;
  if (n >= N_NODES) return;
  float* row = node + (size_t)n * 288;
  const float S = 0.25f;                 // AVG^{-1/2} = 16^{-1/2}
  const float x0 = row[lane] * S;
  float x1[3], x2[5];
  #pragma unroll
  for (int c = 0; c < 3; ++c) x1[c] = row[32 + lane * 3 + c] * S;
  #pragma unroll
  for (int c = 0; c < 5; ++c) x2[c] = row[128 + lane * 5 + c] * S;

  float s = x0;
  #pragma unroll
  for (int off = 16; off > 0; off >>= 1) s += __shfl_xor(s, off, 32);
  const float m0 = s * (1.0f / 32.0f);
  const float d = x0 - m0;
  float v = d * d;
  #pragma unroll
  for (int off = 16; off > 0; off >>= 1) v += __shfl_xor(v, off, 32);
  const float n0n = d * rsqrtf(v * (1.0f / 32.0f) + EPSF) * ln_w0[lane] + ln_b0[lane];

  float p = 0.0f, q = 0.0f;
  #pragma unroll
  for (int c = 0; c < 3; ++c) p += x1[c] * x1[c];
  #pragma unroll
  for (int c = 0; c < 5; ++c) q += x2[c] * x2[c];
  p = p * (1.0f / 3.0f) + q * (1.0f / 5.0f);
  #pragma unroll
  for (int off = 16; off > 0; off >>= 1) p += __shfl_xor(p, off, 32);
  const float inv = rsqrtf(p * (1.0f / 64.0f) + EPSF);

  row[lane] = n0n;
  const float g1 = ln_w1[lane] * inv;
  const float g2 = ln_w1[32 + lane] * inv;
  #pragma unroll
  for (int c = 0; c < 3; ++c) row[32 + lane * 3 + c] = x1[c] * g1;
  #pragma unroll
  for (int c = 0; c < 5; ++c) row[128 + lane * 5 + c] = x2[c] * g2;
}

// ---------------------------------------------------------------------------
extern "C" void kernel_launch(void* const* d_in, const int* in_sizes, int n_in,
                              void* d_out, int out_size, void* d_ws, size_t ws_size,
                              hipStream_t stream) {
  (void)in_sizes; (void)n_in; (void)out_size; (void)ws_size;
  const int*   edge_index   = (const int*)d_in[0];
  const float* edge_sh      = (const float*)d_in[3];
  const float* edge_length  = (const float*)d_in[4];
  const float* edge_one_hot = (const float*)d_in[5];
  const float* bessel_w     = (const float*)d_in[6];
  const float* W1   = (const float*)d_in[7];
  const float* W2   = (const float*)d_in[8];
  const float* W3   = (const float*)d_in[9];
  const float* We   = (const float*)d_in[10];
  const float* ln_w0 = (const float*)d_in[11];
  const float* ln_b0 = (const float*)d_in[12];
  const float* ln_w1 = (const float*)d_in[13];

  float* out       = (float*)d_out;
  float* out_lat   = out;                                   // E x 128
  float* out_node  = out_lat + (size_t)E_EDGES * 128;       // N x 288
  float* out_edgef = out_node + (size_t)N_NODES * 288;      // E x 288
  float* out_cut   = out_edgef + (size_t)E_EDGES * 288;     // E

  _Float16* wts = (_Float16*)d_ws;                          // 128 KB f16 weights

  prep_weights<<<256, 256, 0, stream>>>(W1, W2, W3, We, wts);
  hipMemsetAsync(out_node, 0, (size_t)N_NODES * 288 * sizeof(float), stream);

  edge_mlp_wmma<<<E_EDGES / 128, 256, SHMEM_BYTES, stream>>>(
      edge_index, edge_sh, edge_length, edge_one_hot, bessel_w, wts,
      out_lat, out_node, out_edgef, out_cut);

  node_norm<<<(N_NODES + 7) / 8, 256, 0, stream>>>(out_node, ln_w0, ln_b0, ln_w1);
}
